// GPT2CrossAttention_16106127360193
// MI455X (gfx1250) — matrix operand order repaired and checked by reading, not compiled, not run on hardware
//
#include <hip/hip_runtime.h>
#include <hip/hip_bf16.h>

// ---------------------------------------------------------------------------
// GPT-2 cross attention forward for MI455X (gfx1250, wave32, WMMA).
//   qp = q @ Wq^T + bq ; kp = ... ; vp = ...          (GEMM, bf16 out)
//   attn = flash_softmax(qp kp^T * scale) vp          (fused, bf16 out)
//   out  = attn @ Wo^T + bo                           (GEMM, fp32 out)
// All matmuls on v_wmma_f32_16x16x32_bf16. Intermediates bf16 in d_ws (32MB),
// fully L2-resident (192MB L2). ~52 GFLOP / ~80MB -> bandwidth bound at
// 23.3 TB/s, so: bf16 everywhere off-chip, fused softmax, software-pipelined
// global->LDS staging, fully unrolled register loops (no v_movrel).
// ---------------------------------------------------------------------------

typedef __attribute__((ext_vector_type(16))) __bf16 v16bf;
typedef __attribute__((ext_vector_type(8)))  __bf16 v8bf;
typedef __attribute__((ext_vector_type(8)))  float  v8f;

__device__ __forceinline__ __bf16 f2bf(float x) {
    // round-to-nearest-even fp32 -> bf16
    unsigned u = __builtin_bit_cast(unsigned, x);
    u += 0x7FFFu + ((u >> 16) & 1u);
    unsigned short s = (unsigned short)(u >> 16);
    return __builtin_bit_cast(__bf16, s);
}

// Gather a v16bf WMMA operand from two contiguous 8-element (16B) chunks.
__device__ __forceinline__ v16bf load_op16(const __bf16* p0, const __bf16* p1) {
    v8bf lo = *(const v8bf*)p0;
    v8bf hi = *(const v8bf*)p1;
    return __builtin_shufflevector(lo, hi, 0,1,2,3,4,5,6,7,8,9,10,11,12,13,14,15);
}

__device__ __forceinline__ v8f wmma_bf16(v16bf a, v16bf b, v8f c) {
    return __builtin_amdgcn_wmma_f32_16x16x32_bf16(
        /*neg_a=*/false, a, /*neg_b=*/false, b,
        /*c_mod=*/(short)0, c, /*reuse_a=*/false, /*reuse_b=*/false);
}

// ---------------------------------------------------------------------------
// GEMM:  Y[M,N] = X[M,K] @ W[N,K]^T + bias[N]
// Block tile 128(M) x 64(N), K-step 32. 256 threads = 8 waves in a 4x2 grid,
// each wave computes a 32x32 sub-tile (2x2 WMMA accumulators). Global loads
// for tile k+1 are issued before the WMMAs of tile k (register double buffer).
// ---------------------------------------------------------------------------
template <typename TIN, typename TOUT>
__global__ __launch_bounds__(256)
void gemm_bias_kernel(const TIN* __restrict__ X, const float* __restrict__ W,
                      const float* __restrict__ bias, TOUT* __restrict__ Y,
                      int M, int N, int K)
{
    constexpr int BM = 128, BN = 64, BK = 32;
    __shared__ __bf16 As[BM][40];   // [m][k], 80B rows: 16B aligned, conflict-free
    __shared__ __bf16 Bs[BN][40];   // [n][k]

    const int tid   = threadIdx.x;
    const int lane  = tid & 31;
    const int wave  = tid >> 5;
    const int waveM = wave >> 1;          // 0..3
    const int waveN = wave & 1;           // 0..1
    const int m0    = blockIdx.y * BM;
    const int n0    = blockIdx.x * BN;

    v8f acc[2][2] = {};

    const int a_row = lane & 15;          // A operand: M = lane&15
    const int a_kb  = (lane >> 4) * 8;    // K sub-offset {0,8}
    const int b_col = lane & 15;          // B operand: N = lane&15
    const int b_kb  = (lane >> 4) * 8;    // K sub-offset {0,8}

    // ---- register staging buffers (double buffer vs LDS tile) ----
    float4 raf[4];      // A tile, fp32 input path
    v8bf   rab[2];      // A tile, bf16 input path
    float4 rbf[2];      // B tile (weights, fp32)

    auto load_tile = [&](int k0) {
        if constexpr (sizeof(TIN) == 4) {
#pragma unroll
            for (int i = 0; i < 4; ++i) {
                int idx = tid + i * 256;              // 1024 float4 chunks
                int r = idx >> 3, c4 = idx & 7;
                raf[i] = *(const float4*)((const float*)X + (size_t)(m0 + r) * K + k0 + c4 * 4);
            }
        } else {
#pragma unroll
            for (int i = 0; i < 2; ++i) {
                int idx = tid + i * 256;              // 512 x 8-elem chunks
                int r = idx >> 2, c8 = idx & 3;
                rab[i] = *(const v8bf*)((const __bf16*)X + (size_t)(m0 + r) * K + k0 + c8 * 8);
            }
        }
#pragma unroll
        for (int i = 0; i < 2; ++i) {
            int idx = tid + i * 256;                  // 512 float4 chunks
            int r = idx >> 3, c4 = idx & 7;
            rbf[i] = *(const float4*)(W + (size_t)(n0 + r) * K + k0 + c4 * 4);
        }
    };

    auto stage_tile = [&]() {
        if constexpr (sizeof(TIN) == 4) {
#pragma unroll
            for (int i = 0; i < 4; ++i) {
                int idx = tid + i * 256;
                int r = idx >> 3, c4 = idx & 7;
                __bf16* d = &As[r][c4 * 4];
                d[0] = f2bf(raf[i].x); d[1] = f2bf(raf[i].y);
                d[2] = f2bf(raf[i].z); d[3] = f2bf(raf[i].w);
            }
        } else {
#pragma unroll
            for (int i = 0; i < 2; ++i) {
                int idx = tid + i * 256;
                int r = idx >> 2, c8 = idx & 3;
                *(v8bf*)&As[r][c8 * 8] = rab[i];
            }
        }
#pragma unroll
        for (int i = 0; i < 2; ++i) {
            int idx = tid + i * 256;
            int r = idx >> 3, c4 = idx & 7;
            __bf16* d = &Bs[r][c4 * 4];
            d[0] = f2bf(rbf[i].x); d[1] = f2bf(rbf[i].y);
            d[2] = f2bf(rbf[i].z); d[3] = f2bf(rbf[i].w);
        }
    };

    load_tile(0);                                     // prologue

    for (int k0 = 0; k0 < K; k0 += BK) {
        __syncthreads();                              // prior reads of LDS done
        stage_tile();
        __syncthreads();

        if (k0 + BK < K) load_tile(k0 + BK);          // prefetch next tile

        v16bf bfrag[2];
#pragma unroll
        for (int tn = 0; tn < 2; ++tn) {
            const __bf16* p = &Bs[waveN * 32 + tn * 16 + b_col][b_kb];
            bfrag[tn] = load_op16(p, p + 16);         // K {kb..kb+7, 16+kb..16+kb+7}
        }
#pragma unroll
        for (int tm = 0; tm < 2; ++tm) {
            const __bf16* p = &As[waveM * 32 + tm * 16 + a_row][a_kb];
            v16bf afrag = load_op16(p, p + 16);       // K {kb..kb+7, 16+kb..16+kb+7}
#pragma unroll
            for (int tn = 0; tn < 2; ++tn)
                acc[tm][tn] = wmma_bf16(afrag, bfrag[tn], acc[tm][tn]);
        }
    }

    // ---- epilogue: bias + store (C layout: row = v + 8*(lane>>4), col = lane&15)
    const int hlf = lane >> 4;
#pragma unroll
    for (int tm = 0; tm < 2; ++tm)
#pragma unroll
        for (int tn = 0; tn < 2; ++tn) {
            int col = n0 + waveN * 32 + tn * 16 + (lane & 15);
            float bv = bias[col];
            int rbase = m0 + waveM * 32 + tm * 16 + hlf * 8;
#pragma unroll
            for (int v = 0; v < 8; ++v) {
                float val = acc[tm][tn][v] + bv;
                if constexpr (sizeof(TOUT) == 2)
                    ((__bf16*)Y)[(size_t)(rbase + v) * N + col] = f2bf(val);
                else
                    ((float*)Y)[(size_t)(rbase + v) * N + col] = val;
            }
        }
}

// ---------------------------------------------------------------------------
// Flash attention: block = (batch*head, 128 queries). Each of 8 waves owns 16
// queries; Q operands live in registers for the whole Tk loop. 32-key tiles:
// S = Q K^T (2 WMMA per 16-key column tile), online softmax via shfl_xor over
// the 16-lane column groups (fully unrolled -> no movrel), P restaged through
// wave-private LDS into the A-operand layout, O += P V (4 WMMA over d=64).
// K/V tiles are register-double-buffered against the LDS stage.
// ---------------------------------------------------------------------------
#define N_HEADS 16
#define HS 64

__global__ __launch_bounds__(256)
void attention_kernel(const __bf16* __restrict__ Qp, const __bf16* __restrict__ Kp,
                      const __bf16* __restrict__ Vp, __bf16* __restrict__ O,
                      int Tq, int Tk, float scale)
{
    constexpr int BQ = 128, BKV = 32;
    __shared__ __bf16 Ks[BKV][72];      // [key][d]  (64 + pad)
    __shared__ __bf16 Vs[HS][40];       // [d][key]  (transposed V, 32 + pad)
    __shared__ __bf16 Ps[8][16][40];    // per-wave P tile [q][key]

    const int tid  = threadIdx.x, lane = tid & 31, wave = tid >> 5;
    const int bh   = blockIdx.y;
    const int b    = bh / N_HEADS, h = bh % N_HEADS;
    const int q0   = blockIdx.x * BQ;
    const size_t C = (size_t)N_HEADS * HS;

    const __bf16* qbase = Qp + (size_t)b * Tq * C + (size_t)h * HS;
    const __bf16* kbase = Kp + (size_t)b * Tk * C + (size_t)h * HS;
    const __bf16* vbase = Vp + (size_t)b * Tk * C + (size_t)h * HS;

    const int col  = lane & 15;
    const int hlf  = lane >> 4;
    const int kb8  = hlf * 8;           // A-operand K sub-offset
    const int kb16 = hlf * 8;           // B-operand K sub-offset

    // Q operands (A layout), kept in registers: 2 chunks over d = 64
    const int qrow = q0 + wave * 16 + col;
    v16bf qfrag[2];
#pragma unroll
    for (int c = 0; c < 2; ++c) {
        const __bf16* p = qbase + (size_t)qrow * C + c * 32 + kb8;
        qfrag[c] = load_op16(p, p + 16);
    }

    float mrow[8], lrow[8];
    v8f acc_o[4] = {};
#pragma unroll
    for (int v = 0; v < 8; ++v) { mrow[v] = -1e30f; lrow[v] = 0.0f; }

    // ---- K/V register staging: one 16B chunk each per thread ----
    const int skey = tid >> 3;          // 0..31
    const int sd8  = (tid & 7) * 8;     // 0..56
    v8bf kreg, vreg;
    auto load_kv = [&](int t0) {
        kreg = *(const v8bf*)(kbase + (size_t)(t0 + skey) * C + sd8);
        vreg = *(const v8bf*)(vbase + (size_t)(t0 + skey) * C + sd8);
    };
    load_kv(0);                                       // prologue

    for (int t0 = 0; t0 < Tk; t0 += BKV) {
        __syncthreads();
        *(v8bf*)&Ks[skey][sd8] = kreg;                // [key][d]
#pragma unroll
        for (int j = 0; j < 8; ++j)                   // transpose: [d][key]
            Vs[sd8 + j][skey] = vreg[j];
        __syncthreads();

        if (t0 + BKV < Tk) load_kv(t0 + BKV);         // prefetch next tile

        // ---- scores: two 16-key tiles, K-reduce over d in 2 WMMA steps ----
        v8f s[2] = {};
#pragma unroll
        for (int kt = 0; kt < 2; ++kt) {
            const __bf16* p0 = &Ks[kt * 16 + col][kb16];       // d chunk 0
            const __bf16* p1 = &Ks[kt * 16 + col][32 + kb16];  // d chunk 1
            v16bf b0 = load_op16(p0, p0 + 16);
            v16bf b1 = load_op16(p1, p1 + 16);
            s[kt] = wmma_bf16(qfrag[0], b0, s[kt]);
            s[kt] = wmma_bf16(qfrag[1], b1, s[kt]);
        }

        // ---- online softmax (rows live across the 16 lanes of each half) ----
#pragma unroll
        for (int v = 0; v < 8; ++v) {
            float s0 = s[0][v] * scale, s1 = s[1][v] * scale;
            float mx = fmaxf(s0, s1);
            mx = fmaxf(mx, __shfl_xor(mx, 1, 32));
            mx = fmaxf(mx, __shfl_xor(mx, 2, 32));
            mx = fmaxf(mx, __shfl_xor(mx, 4, 32));
            mx = fmaxf(mx, __shfl_xor(mx, 8, 32));
            float mnew  = fmaxf(mrow[v], mx);
            float alpha = __expf(mrow[v] - mnew);
            float p0 = __expf(s0 - mnew);
            float p1 = __expf(s1 - mnew);
            float ps = p0 + p1;
            ps += __shfl_xor(ps, 1, 32);
            ps += __shfl_xor(ps, 2, 32);
            ps += __shfl_xor(ps, 4, 32);
            ps += __shfl_xor(ps, 8, 32);
            lrow[v] = lrow[v] * alpha + ps;
            mrow[v] = mnew;
#pragma unroll
            for (int t = 0; t < 4; ++t) acc_o[t][v] *= alpha;
            int r = v + hlf * 8;                       // C-layout row -> LDS
            Ps[wave][r][col]      = f2bf(p0);
            Ps[wave][r][16 + col] = f2bf(p1);
        }
        // same-wave LDS store->load ordering (wave-private buffer, no barrier)
        asm volatile("s_wait_dscnt 0" ::: "memory");

        // ---- O += P @ V : 4 WMMA over the 64-wide d dimension ----
        {
            const __bf16* pp = &Ps[wave][col][kb8];
            v16bf pfrag = load_op16(pp, pp + 16);      // A layout 16x32
#pragma unroll
            for (int t = 0; t < 4; ++t) {
                const __bf16* pv = &Vs[t * 16 + col][kb16];
                v16bf vfrag = load_op16(pv, pv + 16);
                acc_o[t] = wmma_bf16(pfrag, vfrag, acc_o[t]);
            }
        }
    }

    // ---- normalize + store (bf16, [b][q][h*HS + d]) ----
    __bf16* obase = O + (size_t)b * Tq * C + (size_t)h * HS;
#pragma unroll
    for (int v = 0; v < 8; ++v) {
        float inv = 1.0f / lrow[v];
        int r = q0 + wave * 16 + hlf * 8 + v;
#pragma unroll
        for (int t = 0; t < 4; ++t)
            obase[(size_t)r * C + t * 16 + col] = f2bf(acc_o[t][v] * inv);
    }
}

// ---------------------------------------------------------------------------
extern "C" void kernel_launch(void* const* d_in, const int* in_sizes, int n_in,
                              void* d_out, int out_size, void* d_ws, size_t ws_size,
                              hipStream_t stream)
{
    const float* q  = (const float*)d_in[0];
    const float* k  = (const float*)d_in[1];
    const float* v  = (const float*)d_in[2];
    const float* Wq = (const float*)d_in[3];
    const float* bq = (const float*)d_in[4];
    const float* Wk = (const float*)d_in[5];
    const float* bk = (const float*)d_in[6];
    const float* Wv = (const float*)d_in[7];
    const float* bv = (const float*)d_in[8];
    const float* Wo = (const float*)d_in[9];
    const float* bo = (const float*)d_in[10];
    float* out = (float*)d_out;

    const int B = 4, T = 1024, C = 1024;
    const int M = B * T;                       // 4096
    const size_t elems = (size_t)M * C;        // 4M bf16 per buffer

    __bf16* qp = (__bf16*)d_ws;
    __bf16* kp = qp + elems;
    __bf16* vp = kp + elems;
    __bf16* ap = vp + elems;

    dim3 blk(256);
    dim3 gGemm(C / 64, M / 128);               // 16 x 32
    dim3 gAttn(T / 128, B * N_HEADS);          // 8 x 64

    hipLaunchKernelGGL((gemm_bias_kernel<float, __bf16>), gGemm, blk, 0, stream,
                       q, Wq, bq, qp, M, C, C);
    hipLaunchKernelGGL((gemm_bias_kernel<float, __bf16>), gGemm, blk, 0, stream,
                       k, Wk, bk, kp, M, C, C);
    hipLaunchKernelGGL((gemm_bias_kernel<float, __bf16>), gGemm, blk, 0, stream,
                       v, Wv, bv, vp, M, C, C);
    hipLaunchKernelGGL(attention_kernel, gAttn, blk, 0, stream,
                       qp, kp, vp, ap, T, T, 0.125f /* 64^-0.5 */);
    hipLaunchKernelGGL((gemm_bias_kernel<__bf16, float>), gGemm, blk, 0, stream,
                       ap, Wo, bo, out, M, C, C);
}
